// PairwiseBiasAttention_76957224009944
// MI455X (gfx1250) — matrix operand order repaired and checked by reading, not compiled, not run on hardware
//
#include <hip/hip_runtime.h>
#include <hip/hip_bf16.h>

// MI455X / gfx1250, wave32. bf16 WMMA + async global->LDS staging (ASYNCcnt).
// One-time prep kernels pre-convert/transpose operands so GEMM tile staging is
// pure async b128 copies. Flash pass streams pair_repr (512 MB) exactly once.

typedef __bf16 bf16;
typedef __attribute__((ext_vector_type(16))) __bf16    bf16x16;
typedef __attribute__((ext_vector_type(8)))  float     f32x8;
typedef __attribute__((ext_vector_type(8)))  unsigned  u32x8;

union Frag { u32x8 u; bf16x16 h; };

__device__ __forceinline__ f32x8 wmma_bf16(bf16x16 a, bf16x16 b, f32x8 c) {
  return __builtin_amdgcn_wmma_f32_16x16x32_bf16(false, a, false, b, (short)0, c, false, false);
}

__device__ __forceinline__ unsigned pack2(float x, float y) {  // -> v_cvt_pk_bf16_f32
  union { bf16 h[2]; unsigned u; } t;
  t.h[0] = (bf16)x; t.h[1] = (bf16)y; return t.u;
}

// 16-byte async copy global -> LDS (GLOBAL_LOAD_ASYNC_TO_LDS_B128, ASYNCcnt).
// LDS generic pointers carry the LDS byte offset in the low 32 bits.
__device__ __forceinline__ void async_copy16(const bf16* g, bf16* l) {
  asm volatile("global_load_async_to_lds_b128 %0, %1, off"
               :: "v"((unsigned)(uintptr_t)l),
                  "v"((unsigned long long)(uintptr_t)g)
               : "memory");
}
__device__ __forceinline__ void wait_async0() {
  asm volatile("s_wait_asynccnt 0x0" ::: "memory");
}

#define NB 2
#define NL 1024
#define NE 512
#define NH 8
#define ND 64
#define NP 64
#define NSEG 2
#define LOG2E 1.44269504f

// ---------------------------------------------------------------------------
// Prep: f32 -> bf16 elementwise (seq_repr)
// ---------------------------------------------------------------------------
__global__ __launch_bounds__(256)
void cvt_f32_bf16(const float* __restrict__ in, bf16* __restrict__ outp) {
  size_t i = ((size_t)blockIdx.x * 256 + threadIdx.x) * 4;
  float4 t = *(const float4*)&in[i];
  uint2 r; r.x = pack2(t.x, t.y); r.y = pack2(t.z, t.w);
  *(uint2*)&outp[i] = r;
}

// ---------------------------------------------------------------------------
// Prep: f32 [R][C] -> bf16 [C][R] (LDS 32x32 tile transpose, coalesced both ways)
// ---------------------------------------------------------------------------
__global__ __launch_bounds__(256)
void tr_cvt(const float* __restrict__ in, bf16* __restrict__ outp, int R, int C) {
  __shared__ float tile[32][33];
  const int tx = threadIdx.x & 31, ty = threadIdx.x >> 5;   // 32 x 8
  #pragma unroll
  for (int p = 0; p < 4; ++p)
    tile[ty + p * 8][tx] = in[(size_t)(blockIdx.y * 32 + ty + p * 8) * C + blockIdx.x * 32 + tx];
  __syncthreads();
  #pragma unroll
  for (int p = 0; p < 4; ++p)
    outp[(size_t)(blockIdx.x * 32 + ty + p * 8) * R + blockIdx.y * 32 + tx] =
        (bf16)tile[tx][ty + p * 8];
}

// ---------------------------------------------------------------------------
// Kernel 1: qkv = seqb[2048,512] @ WqkvT[1536,512]^T -> q,k [B,H,L,64], vT [B,H,64,L]
// 8 waves (2x4), tile 64(M) x 128(N), BK=32, double-buffered async LDS staging.
// ---------------------------------------------------------------------------
__global__ __launch_bounds__(256)
void qkv_gemm(const bf16* __restrict__ seqb, const bf16* __restrict__ WqkvT,
              bf16* __restrict__ qws, bf16* __restrict__ kws, bf16* __restrict__ vTws) {
  __shared__ bf16 As[2][64][40];     // [buf][m][k], row stride 80 B
  __shared__ bf16 BsT[2][128][40];   // [buf][n][k]
  const int tid = threadIdx.x;
  const int lane = tid & 31, w = tid >> 5;
  const int lm = lane & 15, lg = lane >> 4;
  const int wm = w & 1, wn = w >> 1;
  const int bm = blockIdx.x * 64, bn = blockIdx.y * 128;

  const int arow = tid >> 2, aq = tid & 3;
  auto issue = [&](int bb, int k0) {
    async_copy16(&seqb[(size_t)(bm + arow) * 512 + k0 + aq * 8], &As[bb][arow][aq * 8]);
    #pragma unroll
    for (int p = 0; p < 2; ++p) {
      int idx = tid + p * 256;
      int r = idx >> 2, q = idx & 3;
      async_copy16(&WqkvT[(size_t)(bn + r) * 512 + k0 + q * 8], &BsT[bb][r][q * 8]);
    }
  };

  f32x8 acc[2][2] = {};
  issue(0, 0);
  wait_async0();
  __syncthreads();

  for (int s = 0; s < 16; ++s) {
    const int buf = s & 1;
    if (s + 1 < 16) issue(buf ^ 1, (s + 1) * 32);

    Frag af[2], bfv[2];
    #pragma unroll
    for (int mt = 0; mt < 2; ++mt) {
      const unsigned* ar = (const unsigned*)&As[buf][wm * 32 + mt * 16 + lm][0];
      #pragma unroll
      for (int v = 0; v < 8; ++v)
        af[mt].u[v] = ar[((v >> 2) << 3) + (lg << 2) + (v & 3)];
    }
    #pragma unroll
    for (int nt = 0; nt < 2; ++nt) {
      const unsigned* br = (const unsigned*)&BsT[buf][wn * 32 + nt * 16 + lm][0];
      #pragma unroll
      for (int v = 0; v < 8; ++v)
        bfv[nt].u[v] = br[((v >> 2) << 3) + (lg << 2) + (v & 3)];
    }
    #pragma unroll
    for (int mt = 0; mt < 2; ++mt)
      #pragma unroll
      for (int nt = 0; nt < 2; ++nt)
        acc[mt][nt] = wmma_bf16(af[mt].h, bfv[nt].h, acc[mt][nt]);

    wait_async0();          // next buffer's async writes landed
    __syncthreads();        // publish; also protects WAR on buf^1
  }

  #pragma unroll
  for (int mt = 0; mt < 2; ++mt)
    #pragma unroll
    for (int nt = 0; nt < 2; ++nt)
      #pragma unroll
      for (int r = 0; r < 8; ++r) {
        int gm = bm + wm * 32 + mt * 16 + r + (lg << 3);
        int gn = bn + wn * 32 + nt * 16 + lm;
        int which = gn >> 9, nn = gn & 511;
        int h = nn >> 6, d = nn & 63;
        int b = gm >> 10, l = gm & 1023;
        bf16 val = (bf16)acc[mt][nt][r];
        if (which == 0)      qws[((size_t)(b * NH + h) * NL + l) * ND + d] = val;
        else if (which == 1) kws[((size_t)(b * NH + h) * NL + l) * ND + d] = val;
        else                 vTws[((size_t)(b * NH + h) * ND + d) * NL + l] = val;
      }
}

// ---------------------------------------------------------------------------
// Kernel 2: fused pair-bias + flash attention (split-KV partials).
// Grid = NSEG * B * 64 i-tiles. 8 waves = 8 heads (pair_repr read exactly once).
// All exponentials in base-2: log2(e) folded into scale and W_bias fragments.
// ---------------------------------------------------------------------------
__global__ __launch_bounds__(256)
void attn_fused(const float* __restrict__ pair, const float* __restrict__ Wbias,
                const bf16* __restrict__ qws, const bf16* __restrict__ kws,
                const bf16* __restrict__ vTws, float* __restrict__ Opart,
                float* __restrict__ mpart, float* __restrict__ lpart) {
  __shared__ float bias_lds[16][32][8];   // [i][j][h], holds bias*log2e
  __shared__ bf16  p_lds[8][16][32];      // per-wave P tile
  const int tid = threadIdx.x, lane = tid & 31, w = tid >> 5;   // w == head
  const int lm = lane & 15, lg = lane >> 4;
  const int seg = blockIdx.x >> 7;
  const int b   = (blockIdx.x >> 6) & 1;
  const int it  = blockIdx.x & 63;
  const int i0  = it << 4;
  const int jlo = seg * (NL / NSEG), jhi = jlo + NL / NSEG;

  // W_bias (scaled by log2e) as B-frags: (k=p, n=h), n padded 8->16 with zeros
  Frag wbf[2];
  #pragma unroll
  for (int kk = 0; kk < 2; ++kk)
    #pragma unroll
    for (int v = 0; v < 8; ++v) {
      int p = kk * 32 + ((v >> 2) << 4) + (lg << 3) + ((v & 3) << 1);
      float a0 = (lm < 8) ? Wbias[p * NH + lm]       * LOG2E : 0.f;
      float a1 = (lm < 8) ? Wbias[(p + 1) * NH + lm] * LOG2E : 0.f;
      wbf[kk].u[v] = pack2(a0, a1);
    }

  // Q fragments: 8 consecutive bf16 per 4-VGPR group -> uint4 loads
  Frag qf[2];
  {
    const uint4* q4 = (const uint4*)(qws + ((size_t)(b * NH + w) * NL + (i0 + lm)) * ND);
    #pragma unroll
    for (int kk = 0; kk < 2; ++kk)
      #pragma unroll
      for (int g = 0; g < 2; ++g) {
        uint4 t = q4[kk * 4 + g * 2 + lg];
        qf[kk].u[4*g+0] = t.x; qf[kk].u[4*g+1] = t.y;
        qf[kk].u[4*g+2] = t.z; qf[kk].u[4*g+3] = t.w;
      }
  }

  float m_i[8], l_i[8];
  f32x8 acc_o[4] = {};
  #pragma unroll
  for (int r = 0; r < 8; ++r) { m_i[r] = -1e30f; l_i[r] = 0.f; }
  const float scale = 0.125f * LOG2E;   // 1/sqrt(64) * log2(e)

  for (int j0 = jlo; j0 < jhi; j0 += 32) {
    __syncthreads();

    if (j0 + 32 < jhi) {                  // prefetch next pair tile (global_prefetch_b8)
      int pi = tid >> 4, pj = (tid & 15) << 1;
      __builtin_prefetch(pair + (((size_t)b * NL + (i0 + pi)) * NL + (j0 + 32 + pj)) * NP, 0, 1);
    }

    // ---- cooperative pair-bias: wave w handles 4 of 32 M-tiles ----
    #pragma unroll
    for (int tt = 0; tt < 4; ++tt) {
      int t  = w * 4 + tt;
      int ii = t >> 1, jh = (t & 1) << 4;
      const float4* p4 = (const float4*)(pair + (((size_t)b * NL + (i0 + ii)) * NL + (j0 + jh + lm)) * NP);
      f32x8 bacc = {};
      #pragma unroll
      for (int kk = 0; kk < 2; ++kk) {
        Frag af;
        #pragma unroll
        for (int g = 0; g < 2; ++g) {
          float4 a = p4[kk * 8 + g * 4 + lg * 2];
          float4 c = p4[kk * 8 + g * 4 + lg * 2 + 1];
          af.u[4*g+0] = pack2(a.x, a.y); af.u[4*g+1] = pack2(a.z, a.w);
          af.u[4*g+2] = pack2(c.x, c.y); af.u[4*g+3] = pack2(c.z, c.w);
        }
        bacc = wmma_bf16(af.h, wbf[kk].h, bacc);
      }
      #pragma unroll
      for (int r = 0; r < 8; ++r)
        if (lm < 8) bias_lds[ii][jh + r + (lg << 3)][lm] = bacc[r];
    }
    __syncthreads();

    // ---- S' = log2e*(scale0*Q K^T + bias) per wave / head ----
    f32x8 sv[2];
    #pragma unroll
    for (int jn = 0; jn < 2; ++jn) {
      const uint4* k4 = (const uint4*)(kws + ((size_t)(b * NH + w) * NL + (j0 + jn * 16 + lm)) * ND);
      f32x8 s = {};
      #pragma unroll
      for (int kk = 0; kk < 2; ++kk) {
        Frag kf;
        uint4 t0 = k4[kk * 4 + lg];
        uint4 t1 = k4[kk * 4 + 2 + lg];
        kf.u[0] = t0.x; kf.u[1] = t0.y; kf.u[2] = t0.z; kf.u[3] = t0.w;
        kf.u[4] = t1.x; kf.u[5] = t1.y; kf.u[6] = t1.z; kf.u[7] = t1.w;
        s = wmma_bf16(qf[kk].h, kf.h, s);
      }
      #pragma unroll
      for (int r = 0; r < 8; ++r)
        sv[jn][r] = s[r] * scale + bias_lds[r + (lg << 3)][jn * 16 + lm][w];
    }

    // ---- online softmax in base-2 (rows replicated across 16-lane halves) ----
    float pw[2][8];
    #pragma unroll
    for (int r = 0; r < 8; ++r) {
      float mx = fmaxf(sv[0][r], sv[1][r]);
      #pragma unroll
      for (int off = 1; off < 16; off <<= 1) mx = fmaxf(mx, __shfl_xor(mx, off, 32));
      float mnew  = fmaxf(m_i[r], mx);
      float alpha = exp2f(m_i[r] - mnew);
      float p0 = exp2f(sv[0][r] - mnew);
      float p1 = exp2f(sv[1][r] - mnew);
      float rs = p0 + p1;
      #pragma unroll
      for (int off = 1; off < 16; off <<= 1) rs += __shfl_xor(rs, off, 32);
      l_i[r] = l_i[r] * alpha + rs;
      m_i[r] = mnew;
      #pragma unroll
      for (int dt = 0; dt < 4; ++dt) acc_o[dt][r] *= alpha;
      pw[0][r] = p0; pw[1][r] = p1;
    }
    #pragma unroll
    for (int jn = 0; jn < 2; ++jn)
      #pragma unroll
      for (int r = 0; r < 8; ++r)
        p_lds[w][r + (lg << 3)][jn * 16 + lm] = (bf16)pw[jn][r];

    // ---- O += P * V (P from LDS via u32; V transposed -> contiguous along j) ----
    Frag pf;
    {
      const unsigned* pr = (const unsigned*)&p_lds[w][lm][0];
      #pragma unroll
      for (int v = 0; v < 8; ++v)
        pf.u[v] = pr[((v >> 2) << 3) + (lg << 2) + (v & 3)];
    }
    #pragma unroll
    for (int dt = 0; dt < 4; ++dt) {
      const uint4* v4 = (const uint4*)(vTws + ((size_t)(b * NH + w) * ND + dt * 16 + lm) * NL + j0);
      Frag vf;
      uint4 t0 = v4[lg];
      uint4 t1 = v4[2 + lg];
      vf.u[0] = t0.x; vf.u[1] = t0.y; vf.u[2] = t0.z; vf.u[3] = t0.w;
      vf.u[4] = t1.x; vf.u[5] = t1.y; vf.u[6] = t1.z; vf.u[7] = t1.w;
      acc_o[dt] = wmma_bf16(pf.h, vf.h, acc_o[dt]);
    }
  }

  // ---- write un-normalized partials ----
  const size_t tile = (size_t)(seg * NB + b) * 64 + it;
  #pragma unroll
  for (int dt = 0; dt < 4; ++dt)
    #pragma unroll
    for (int r = 0; r < 8; ++r) {
      int ii = r + (lg << 3);
      Opart[(tile * NH + w) * 1024 + ii * 64 + dt * 16 + lm] = acc_o[dt][r];
    }
  if (lm == 0) {
    #pragma unroll
    for (int r = 0; r < 8; ++r) {
      int ii = r + (lg << 3);
      mpart[tile * 128 + w * 16 + ii] = m_i[r];
      lpart[tile * 128 + w * 16 + ii] = l_i[r];
    }
  }
}

// ---------------------------------------------------------------------------
// Kernel 2b: base-2 log-sum-exp combine of NSEG partials -> ctx bf16 [B,L,512]
// ---------------------------------------------------------------------------
__global__ __launch_bounds__(256)
void attn_combine(const float* __restrict__ Opart, const float* __restrict__ mpart,
                  const float* __restrict__ lpart, bf16* __restrict__ ctx) {
  int idx = blockIdx.x * 256 + threadIdx.x;           // B*L*E = 2^20
  int d = idx & 63, h = (idx >> 6) & 7, i = (idx >> 9) & 1023, b = idx >> 19;
  int it = i >> 4, ii = i & 15;
  size_t t0 = (size_t)(0 * NB + b) * 64 + it;
  size_t t1 = (size_t)(1 * NB + b) * 64 + it;
  float m0 = mpart[t0 * 128 + h * 16 + ii], m1 = mpart[t1 * 128 + h * 16 + ii];
  float l0 = lpart[t0 * 128 + h * 16 + ii], l1 = lpart[t1 * 128 + h * 16 + ii];
  float M  = fmaxf(m0, m1);
  float w0 = exp2f(m0 - M), w1 = exp2f(m1 - M);
  float L  = l0 * w0 + l1 * w1;
  float o0 = Opart[(t0 * NH + h) * 1024 + ii * 64 + d];
  float o1 = Opart[(t1 * NH + h) * 1024 + ii * 64 + d];
  ctx[((size_t)b * NL + i) * NE + h * 64 + d] = (bf16)((o0 * w0 + o1 * w1) / L);
}

// ---------------------------------------------------------------------------
// Kernel 3: out = ctx[2048,512](bf16) @ WoutT[512,512]^T + bout -> f32
// Same double-buffered async staging as kernel 1.
// ---------------------------------------------------------------------------
__global__ __launch_bounds__(256)
void out_gemm(const bf16* __restrict__ ctx, const bf16* __restrict__ WoutT,
              const float* __restrict__ bout, float* __restrict__ out) {
  __shared__ bf16 As[2][64][40];
  __shared__ bf16 BsT[2][128][40];
  const int tid = threadIdx.x;
  const int lane = tid & 31, w = tid >> 5;
  const int lm = lane & 15, lg = lane >> 4;
  const int wm = w & 1, wn = w >> 1;
  const int bm = blockIdx.x * 64, bn = blockIdx.y * 128;

  const int arow = tid >> 2, aq = tid & 3;
  auto issue = [&](int bb, int k0) {
    async_copy16(&ctx[(size_t)(bm + arow) * 512 + k0 + aq * 8], &As[bb][arow][aq * 8]);
    #pragma unroll
    for (int p = 0; p < 2; ++p) {
      int idx = tid + p * 256;
      int r = idx >> 2, q = idx & 3;
      async_copy16(&WoutT[(size_t)(bn + r) * 512 + k0 + q * 8], &BsT[bb][r][q * 8]);
    }
  };

  f32x8 acc[2][2] = {};
  issue(0, 0);
  wait_async0();
  __syncthreads();

  for (int s = 0; s < 16; ++s) {
    const int buf = s & 1;
    if (s + 1 < 16) issue(buf ^ 1, (s + 1) * 32);

    Frag af[2], bfv[2];
    #pragma unroll
    for (int mt = 0; mt < 2; ++mt) {
      const unsigned* ar = (const unsigned*)&As[buf][wm * 32 + mt * 16 + lm][0];
      #pragma unroll
      for (int v = 0; v < 8; ++v)
        af[mt].u[v] = ar[((v >> 2) << 3) + (lg << 2) + (v & 3)];
    }
    #pragma unroll
    for (int nt = 0; nt < 2; ++nt) {
      const unsigned* br = (const unsigned*)&BsT[buf][wn * 32 + nt * 16 + lm][0];
      #pragma unroll
      for (int v = 0; v < 8; ++v)
        bfv[nt].u[v] = br[((v >> 2) << 3) + (lg << 2) + (v & 3)];
    }
    #pragma unroll
    for (int mt = 0; mt < 2; ++mt)
      #pragma unroll
      for (int nt = 0; nt < 2; ++nt)
        acc[mt][nt] = wmma_bf16(af[mt].h, bfv[nt].h, acc[mt][nt]);

    wait_async0();
    __syncthreads();
  }

  #pragma unroll
  for (int mt = 0; mt < 2; ++mt)
    #pragma unroll
    for (int nt = 0; nt < 2; ++nt)
      #pragma unroll
      for (int r = 0; r < 8; ++r) {
        int gm = bm + wm * 32 + mt * 16 + r + (lg << 3);
        int gn = bn + wn * 32 + nt * 16 + lm;
        out[(size_t)gm * 512 + gn] = acc[mt][nt][r] + bout[gn];
      }
}

// ---------------------------------------------------------------------------
extern "C" void kernel_launch(void* const* d_in, const int* in_sizes, int n_in,
                              void* d_out, int out_size, void* d_ws, size_t ws_size,
                              hipStream_t stream) {
  const float* seq   = (const float*)d_in[0];   // [2,1024,512]
  const float* pair  = (const float*)d_in[1];   // [2,1024,1024,64]
  const float* Wqkv  = (const float*)d_in[2];   // [512,1536]
  const float* Wbias = (const float*)d_in[3];   // [64,8]
  const float* Wout  = (const float*)d_in[4];   // [512,512]
  const float* bout  = (const float*)d_in[5];   // [512]
  float* out = (float*)d_out;                   // [2,1024,512] f32

  const size_t qkv_elems = (size_t)NB * NH * NL * ND;    // 1 Mi bf16 each
  bf16* qws   = (bf16*)d_ws;
  bf16* kws   = qws + qkv_elems;
  bf16* vTws  = kws + qkv_elems;                         // transposed [B,H,64,L]
  bf16* ctxws = vTws + qkv_elems;                        // [B,L,512] bf16
  float* Opart = (float*)(ctxws + (size_t)NB * NL * NE); // 8 MB
  float* mpart = Opart + (size_t)NSEG * NB * 64 * NH * 16 * 64;
  float* lpart = mpart + (size_t)NSEG * NB * 64 * 128;
  bf16* seqb   = (bf16*)(lpart + (size_t)NSEG * NB * 64 * 128);  // [2048,512]
  bf16* WqkvT  = seqb  + (size_t)NB * NL * NE;                   // [1536,512]
  bf16* WoutT  = WqkvT + (size_t)NE * 3 * NE;                    // [512,512]

  // one-time operand prep (pure-copy staging afterwards)
  cvt_f32_bf16<<<dim3((NB * NL * NE) / 1024), 256, 0, stream>>>(seq, seqb);
  tr_cvt      <<<dim3(1536 / 32, 512 / 32),   256, 0, stream>>>(Wqkv, WqkvT, 512, 1536);
  tr_cvt      <<<dim3(512 / 32, 512 / 32),    256, 0, stream>>>(Wout, WoutT, 512, 512);

  qkv_gemm    <<<dim3(2048 / 64, 1536 / 128), 256, 0, stream>>>(seqb, WqkvT, qws, kws, vTws);
  attn_fused  <<<dim3(NSEG * NB * 64),        256, 0, stream>>>(pair, Wbias, qws, kws, vTws,
                                                                Opart, mpart, lpart);
  attn_combine<<<dim3((NB * NL * NE) / 256),  256, 0, stream>>>(Opart, mpart, lpart, ctxws);
  out_gemm    <<<dim3(2048 / 64, 512 / 128),  256, 0, stream>>>(ctxws, WoutT, bout, out);
}
